// SpaDE_386547057039
// MI455X (gfx1250) — matrix operand order repaired and checked by reading, not compiled, not run on hardware
//
#include <hip/hip_runtime.h>
#include <cstdint>
#include <cstddef>

// Problem sizes (fixed by the reference)
#define N_BATCH 8192
#define D_DIM   1024
#define L_DIM   8192

typedef __attribute__((ext_vector_type(16))) __bf16 v16bf;
typedef __attribute__((ext_vector_type(8)))  float  v8f;

union V16 { uint4 u[2]; v16bf v; };

// round-to-nearest-even float -> bf16 (as raw u16 in low bits)
__device__ __forceinline__ unsigned int f2bf1(float f) {
  unsigned int u = __float_as_uint(f);
  return (u + 0x7FFFu + ((u >> 16) & 1u)) >> 16;
}

// 8 consecutive fp32 -> 8 packed bf16 (uint4)
__device__ __forceinline__ uint4 cvt8_f32_bf16(const float* p) {
  float4 a = *(const float4*)p;
  float4 b = *(const float4*)(p + 4);
  uint4 r;
  r.x = f2bf1(a.x) | (f2bf1(a.y) << 16);
  r.y = f2bf1(a.z) | (f2bf1(a.w) << 16);
  r.z = f2bf1(b.x) | (f2bf1(b.y) << 16);
  r.w = f2bf1(b.z) | (f2bf1(b.w) << 16);
  return r;
}

// ---- block reductions (256 threads = 8 wave32) ----
__device__ __forceinline__ float blockReduceSum(float v, float* red) {
  #pragma unroll
  for (int o = 16; o > 0; o >>= 1) v += __shfl_xor(v, o, 32);
  int w = threadIdx.x >> 5, l = threadIdx.x & 31;
  __syncthreads();
  if (l == 0) red[w] = v;
  __syncthreads();
  float r = red[0];
  #pragma unroll
  for (int i = 1; i < 8; ++i) r += red[i];
  return r;
}

__device__ __forceinline__ float blockReduceMax(float v, float* red) {
  #pragma unroll
  for (int o = 16; o > 0; o >>= 1) v = fmaxf(v, __shfl_xor(v, o, 32));
  int w = threadIdx.x >> 5, l = threadIdx.x & 31;
  __syncthreads();
  if (l == 0) red[w] = v;
  __syncthreads();
  float r = red[0];
  #pragma unroll
  for (int i = 1; i < 8; ++i) r = fmaxf(r, red[i]);
  return r;
}

// ---- prep: fp32 -> bf16 conversion, optional fused row sum-of-squares ----
__global__ __launch_bounds__(256) void prep_bf16_norm(const float* __restrict__ src,
                                                      unsigned short* __restrict__ dst,
                                                      float* __restrict__ norm, int cols) {
  __shared__ float red[8];
  const int row = blockIdx.x;
  const size_t base = (size_t)row * (size_t)cols;
  float acc = 0.f;
  for (int c = threadIdx.x * 4; c < cols; c += 1024) {
    float4 f = *(const float4*)(src + base + c);
    uint2 p;
    p.x = f2bf1(f.x) | (f2bf1(f.y) << 16);
    p.y = f2bf1(f.z) | (f2bf1(f.w) << 16);
    *(uint2*)(dst + base + c) = p;
    acc += f.x * f.x + f.y * f.y + f.z * f.z + f.w * f.w;
  }
  if (norm != nullptr) {
    float s = blockReduceSum(acc, red);
    if (threadIdx.x == 0) norm[row] = s;
  }
}

// ---- bf16 WMMA GEMM, 128x128 block tile, 256 threads (8 waves) ----
// MODE 0: A = bf16 (ws), epilogue v = softplus(raw) * (2*S - xn[m] - wn[n])
// MODE 1: A = fp32 (z, converted on the fly), epilogue x_hat = S + bias[n]
// B is always row-major [Ndim][Kdim] bf16 (protos / W_dec rows = output cols).
template<int MODE>
__global__ __launch_bounds__(256) void gemm_wmma_bf16(
    const void* __restrict__ Ap, const unsigned short* __restrict__ Bp,
    const float* __restrict__ xn, const float* __restrict__ wn,
    const float* __restrict__ scale_raw, const float* __restrict__ bias,
    float* __restrict__ Cout, int Kdim, int Ndim) {
  __shared__ alignas(16) unsigned short As[128][40];  // +8 halves pad, 80B row stride
  __shared__ alignas(16) unsigned short Bs[128][40];

  const int tid  = threadIdx.x;
  const int lane = tid & 31;
  const int wave = tid >> 5;
  const int wr   = wave & 3;   // 4 M-groups of 32 rows
  const int wc   = wave >> 1 >> 1; // == wave >> 2 : 2 N-groups of 64 cols
  const int bm   = blockIdx.y * 128;
  const int bn   = blockIdx.x * 128;

  v8f vzero = {};
  v8f acc[2][4];
  #pragma unroll
  for (int i = 0; i < 2; ++i)
    #pragma unroll
    for (int j = 0; j < 4; ++j) acc[i][j] = vzero;

  // global->LDS chunk mapping: 512 chunks of 8 halves, 2 per thread
  const int c0row = tid >> 2;        // 0..63 (h=0), +64 for h=1
  const int c0off = (tid & 3) * 8;   // 0,8,16,24

  for (int kt = 0; kt < Kdim; kt += 32) {
    uint4 av[2], bv[2];
    #pragma unroll
    for (int h = 0; h < 2; ++h) {
      const int row = c0row + h * 64;
      const size_t ai = (size_t)(bm + row) * Kdim + kt + c0off;
      const size_t bi = (size_t)(bn + row) * Kdim + kt + c0off;
      if (MODE == 1) av[h] = cvt8_f32_bf16((const float*)Ap + ai);
      else           av[h] = *(const uint4*)((const unsigned short*)Ap + ai);
      bv[h] = *(const uint4*)(Bp + bi);
    }
    if (kt + 32 < Kdim) {  // emits global_prefetch_b8 for the next K tile
      const size_t ai = (size_t)(bm + c0row) * Kdim + kt + 32 + c0off;
      const size_t bi = (size_t)(bn + c0row) * Kdim + kt + 32 + c0off;
      if (MODE == 1) __builtin_prefetch((const float*)Ap + ai, 0, 1);
      else           __builtin_prefetch((const unsigned short*)Ap + ai, 0, 1);
      __builtin_prefetch(Bp + bi, 0, 1);
    }
    __syncthreads();   // previous tile fully consumed
    #pragma unroll
    for (int h = 0; h < 2; ++h) {
      const int row = c0row + h * 64;
      *(uint4*)&As[row][c0off] = av[h];
      *(uint4*)&Bs[row][c0off] = bv[h];
    }
    __syncthreads();   // tile visible to all waves

    // A frag (16x32 bf16): lanes 0-15 -> K {0..7,16..23}; lanes 16-31 -> K {8..15,24..31}
    V16 afr[2], bfr[4];
    const int arow = wr * 32 + (lane & 15);
    const int kbA  = (lane & 16) ? 8 : 0;
    #pragma unroll
    for (int mi = 0; mi < 2; ++mi) {
      const unsigned short* pa = &As[arow + mi * 16][kbA];
      afr[mi].u[0] = *(const uint4*)pa;
      afr[mi].u[1] = *(const uint4*)(pa + 16);
    }
    // B frag (32x16 bf16): lane n holds column n; lanes 0-15 K {0..7,16..23}, lanes 16-31 K {8..15,24..31}
    const int bcol = wc * 64 + (lane & 15);
    const int kbB  = (lane & 16) ? 8 : 0;
    #pragma unroll
    for (int ni = 0; ni < 4; ++ni) {
      const unsigned short* pb = &Bs[bcol + ni * 16][kbB];
      bfr[ni].u[0] = *(const uint4*)pb;
      bfr[ni].u[1] = *(const uint4*)(pb + 16);
    }
    #pragma unroll
    for (int mi = 0; mi < 2; ++mi)
      #pragma unroll
      for (int ni = 0; ni < 4; ++ni)
        acc[mi][ni] = __builtin_amdgcn_wmma_f32_16x16x32_bf16(
            false, afr[mi].v, false, bfr[ni].v, (short)0, acc[mi][ni], false, false);
  }

  float sc = 0.f;
  if (MODE == 0) {
    const float raw = *scale_raw;
    sc = (raw > 20.f) ? raw : log1pf(expf(raw));  // softplus
  }

  // C/D layout: VGPR r -> M = r + 8*(lane>=16), N = lane%16
  const int nloc = lane & 15;
  const int mh   = (lane >> 4) & 1;
  #pragma unroll
  for (int mi = 0; mi < 2; ++mi) {
    #pragma unroll
    for (int ni = 0; ni < 4; ++ni) {
      const int gcol = bn + wc * 64 + ni * 16 + nloc;
      #pragma unroll
      for (int r = 0; r < 8; ++r) {
        const int grow = bm + wr * 32 + mi * 16 + mh * 8 + r;
        const float s = acc[mi][ni][r];
        if (MODE == 0)
          Cout[(size_t)grow * Ndim + gcol] = sc * (2.f * s - xn[grow] - wn[gcol]);
        else
          Cout[(size_t)grow * Ndim + gcol] = s + bias[gcol];
      }
    }
  }
}

// ---- sparsemax over each row of z (length 8192), in place, via bisection on tau ----
__global__ __launch_bounds__(256) void sparsemax_rows(float* __restrict__ z, int Ldim) {
  __shared__ float red[8];
  const int tid = threadIdx.x;
  float* rp = z + (size_t)blockIdx.x * Ldim;

  float v[32];
  #pragma unroll
  for (int i = 0; i < 32; ++i) v[i] = rp[i * 256 + tid];

  float mx = -3.402823e38f;
  #pragma unroll
  for (int i = 0; i < 32; ++i) mx = fmaxf(mx, v[i]);
  mx = blockReduceMax(mx, red);

  // tau in [max-1, max); f(tau) = sum max(v-tau,0) - 1 strictly decreasing
  float lo = mx - 1.f, hi = mx;
  for (int it = 0; it < 32; ++it) {
    const float mid = 0.5f * (lo + hi);
    float s = 0.f;
    #pragma unroll
    for (int i = 0; i < 32; ++i) s += fmaxf(v[i] - mid, 0.f);
    s = blockReduceSum(s, red);
    if (s > 1.f) lo = mid; else hi = mid;   // uniform across the block
  }
  const float tau = 0.5f * (lo + hi);
  #pragma unroll
  for (int i = 0; i < 32; ++i) rp[i * 256 + tid] = fmaxf(v[i] - tau, 0.f);
}

extern "C" void kernel_launch(void* const* d_in, const int* in_sizes, int n_in,
                              void* d_out, int out_size, void* d_ws, size_t ws_size,
                              hipStream_t stream) {
  (void)in_sizes; (void)n_in; (void)out_size; (void)ws_size;
  const float* x      = (const float*)d_in[0];  // [N, D]
  const float* protos = (const float*)d_in[1];  // [L, D]
  const float* sp     = (const float*)d_in[2];  // scalar
  const float* W_dec  = (const float*)d_in[3];  // [D, L]
  const float* b_dec  = (const float*)d_in[4];  // [D]

  float* out  = (float*)d_out;
  float* xhat = out;                                   // [N, D]
  float* zbuf = out + (size_t)N_BATCH * D_DIM;         // [N, L] (logits, then z in place)

  char* w = (char*)d_ws;
  unsigned short* Xb = (unsigned short*)w; w += (size_t)N_BATCH * D_DIM * 2;
  unsigned short* Pb = (unsigned short*)w; w += (size_t)L_DIM   * D_DIM * 2;
  unsigned short* Wb = (unsigned short*)w; w += (size_t)D_DIM   * L_DIM * 2;
  float* xn = (float*)w; w += (size_t)N_BATCH * 4;
  float* wn = (float*)w; w += (size_t)L_DIM * 4;

  // 1) bf16 conversions + fp32 row norms
  prep_bf16_norm<<<N_BATCH, 256, 0, stream>>>(x,      Xb, xn, D_DIM);
  prep_bf16_norm<<<L_DIM,   256, 0, stream>>>(protos, Pb, wn, D_DIM);
  prep_bf16_norm<<<D_DIM,   256, 0, stream>>>(W_dec,  Wb, nullptr, L_DIM);

  // 2) logits v = softplus(raw) * (2 x.p - |x|^2 - |p|^2) -> zbuf
  gemm_wmma_bf16<0><<<dim3(L_DIM / 128, N_BATCH / 128), 256, 0, stream>>>(
      Xb, Pb, xn, wn, sp, nullptr, zbuf, D_DIM, L_DIM);

  // 3) sparsemax in place
  sparsemax_rows<<<N_BATCH, 256, 0, stream>>>(zbuf, L_DIM);

  // 4) x_hat = z @ W_dec^T + b_dec
  gemm_wmma_bf16<1><<<dim3(D_DIM / 128, N_BATCH / 128), 256, 0, stream>>>(
      zbuf, Wb, nullptr, nullptr, nullptr, b_dec, xhat, L_DIM, D_DIM);
}
